// PairFeatures_19765439496935
// MI455X (gfx1250) — matrix operand order repaired and checked by reading, not compiled, not run on hardware
//
#include <hip/hip_runtime.h>
#include <hip/hip_bf16.h>

// ---------------- types ----------------
typedef __attribute__((ext_vector_type(16))) __bf16        v16bf;
typedef __attribute__((ext_vector_type(8)))  float         v8f;
typedef __attribute__((ext_vector_type(4)))  unsigned int  u32x4;

union BfVec {
    v16bf v;
    u32x4 q[2];
    unsigned short u[16];
};

// fp32 -> bf16 via hardware convert (round-to-nearest-even)
static __device__ __forceinline__ unsigned short f2bf(float f) {
    union { __bf16 h; unsigned short u; } cv;
    cv.h = (__bf16)f;
    return cv.u;
}

// A-operand (16x32 bf16): lane m = lane&15; elements e0..7 -> K = (lane>>4)*8 + e,
// elements e8..15 -> K = 16 + (lane>>4)*8 + e  => two contiguous 16B loads.
// Works for global or LDS base pointers (LDS base => ds_load_b128).
static __device__ __forceinline__ void loadA(BfVec& a, const unsigned short* base, int lane) {
    int o = (lane >> 4) * 8;
    a.q[0] = *(const u32x4*)(base + o);
    a.q[1] = *(const u32x4*)(base + o + 16);
}
// B-operand (32x16 bf16): lane n = lane&15 (row of [n][k] repacked matrix);
// elements e0..7 -> K = (lane>>4)*8 + e, e8..15 -> K = 16 + (lane>>4)*8 + e (same as A).
static __device__ __forceinline__ void loadB(BfVec& b, const unsigned short* base, int lane) {
    int o = (lane >> 4) * 8;
    b.q[0] = *(const u32x4*)(base + o);
    b.q[1] = *(const u32x4*)(base + o + 16);
}

static __device__ __forceinline__ v8f wmma_bf16(const BfVec& a, const BfVec& b, v8f c) {
    return __builtin_amdgcn_wmma_f32_16x16x32_bf16(
        /*neg_a=*/false, a.v, /*neg_b=*/false, b.v,
        /*c_mod=*/(short)0, c, /*reuse_a=*/false, /*reuse_b=*/false);
}

// Problem constants
#define NB   2
#define NL   384
#define DM   1024
#define DP   128
#define HW   (NL * NL)          // 147456 spatial per map
#define APAD 136                // padded bf16 row stride in LDS (272B = 68 banks -> conflict-free)

// ---------------- stage 0: projections ui/vj (tiny, VALU) ----------------
__global__ __launch_bounds__(256) void proj_kernel(
    const float* __restrict__ h, const float* __restrict__ Wi, const float* __restrict__ bi,
    const float* __restrict__ Wj, const float* __restrict__ bj,
    unsigned short* __restrict__ ui, unsigned short* __restrict__ vj) {
    __shared__ float hs[DM];
    int bl = blockIdx.x;                       // 0..767 : (b,l) row
    const float* hr = h + (size_t)bl * DM;
    for (int i = threadIdx.x; i < DM; i += 256) hs[i] = hr[i];
    __syncthreads();
    int t = threadIdx.x;
    const float* w = (t < DP) ? (Wi + (size_t)t * DM) : (Wj + (size_t)(t - DP) * DM);
    float acc = (t < DP) ? bi[t] : bj[t - DP];
    for (int i = 0; i < DM; i += 4)
        acc += hs[i] * w[i] + hs[i + 1] * w[i + 1] + hs[i + 2] * w[i + 2] + hs[i + 3] * w[i + 3];
    if (t < DP) ui[bl * DP + t]        = f2bf(acc);
    else        vj[bl * DP + (t - DP)] = f2bf(acc);
}

// ---------------- repacks to bf16 GEMM layouts ----------------
__global__ __launch_bounds__(256) void repack_bilinear_kernel(
    const float* __restrict__ W, unsigned short* __restrict__ Wb) {
    int idx = blockIdx.x * 256 + threadIdx.x;  // i*16384 + j*128 + k
    int k = idx & 127, j = (idx >> 7) & 127, i = idx >> 14;
    Wb[(k * DP + j) * DP + i] = f2bf(W[idx]);
}
__global__ __launch_bounds__(256) void repack_mix_kernel(
    const float* __restrict__ mw, unsigned short* __restrict__ wB) {
    int idx = blockIdx.x * 256 + threadIdx.x;  // ((o*128+i)*3+dy)*3+dx
    int dx = idx % 3, dy = (idx / 3) % 3, i = (idx / 9) % DP, o = idx / (9 * DP);
    wB[o * 1152 + (3 * dy + dx) * DP + i] = f2bf(mw[idx]);
}
__global__ __launch_bounds__(256) void repack_w1_kernel(
    const float* __restrict__ w, unsigned short* __restrict__ o) {
    int idx = blockIdx.x * 256 + threadIdx.x;
    o[idx] = f2bf(w[idx]);
}

// Cooperative stage of a 16-row x 128-col bf16 A tile into padded LDS.
// 256 threads: each thread moves exactly one 16B chunk (16 rows x 16 chunks).
static __device__ __forceinline__ void stageA16x128(
    unsigned short* at, const unsigned short* __restrict__ src, size_t row_stride) {
    int row = threadIdx.x >> 4;
    int cc  = (threadIdx.x & 15) * 8;
    *(u32x4*)(at + row * APAD + cc) = *(const u32x4*)(src + (size_t)row * row_stride + cc);
}

// K=128 dual-accumulator WMMA pipe (breaks the WMMA RAW chain: depth 2 instead of 4)
static __device__ __forceinline__ void gemmK128(
    const unsigned short* arow, const unsigned short* brow, int lane, v8f& c0, v8f& c1) {
    for (int kk = 0; kk < DP; kk += 64) {
        BfVec A0, B0, A1, B1;
        loadA(A0, arow + kk, lane);
        loadB(B0, brow + kk, lane);
        loadA(A1, arow + kk + 32, lane);
        loadB(B1, brow + kk + 32, lane);
        c0 = wmma_bf16(A0, B0, c0);
        c1 = wmma_bf16(A1, B1, c1);
    }
}

// ---------------- stage 1: t[b,k,l,j] = sum_i ui[b,l,i] * Wb[k,j,i] ----------------
// Block: one (b,k,lt) A tile staged in LDS; 8 waves cover jt = 0..7.
__global__ __launch_bounds__(256) void bilinear1_kernel(
    const unsigned short* __restrict__ ui, const unsigned short* __restrict__ Wb,
    unsigned short* __restrict__ t) {
    __shared__ __align__(16) unsigned short at[16 * APAD];
    int bid  = blockIdx.x;                 // 0..6143
    int lt = bid % 24;
    int k  = (bid / 24) % DP;
    int b  = bid / (24 * DP);
    stageA16x128(at, ui + (size_t)(b * NL + lt * 16) * DP, DP);
    __syncthreads();

    int lane = threadIdx.x & 31;
    int jt   = threadIdx.x >> 5;
    const unsigned short* arow = at + (lane & 15) * APAD;
    const unsigned short* brow = Wb + ((k * DP + jt * 16 + (lane & 15)) * DP);
    v8f c0 = {}, c1 = {};
    gemmK128(arow, brow, lane, c0, c1);
    unsigned short* out = t + ((size_t)(b * DP + k) * NL + lt * 16) * DP + jt * 16 + (lane & 15);
    int mb = (lane >> 4) * 8;
    for (int r = 0; r < 8; ++r) out[(mb + r) * DP] = f2bf(c0[r] + c1[r]);
}

// ---------------- stage 2: feat[b,k,l,m] = sum_j t[b,k,l,j] * vj[b,m,j] ----------------
// Block: one (b,k,lt) A tile staged in LDS; 8 waves cover one group of 8 mt tiles.
__global__ __launch_bounds__(256) void pairgemm_kernel(
    const unsigned short* __restrict__ t, const unsigned short* __restrict__ vj,
    float* __restrict__ feat) {
    __shared__ __align__(16) unsigned short at[16 * APAD];
    int bid = blockIdx.x;                  // 0..18431
    int mtg = bid % 3;
    int lt  = (bid / 3) % 24;
    int k   = (bid / 72) % DP;
    int b   = bid / (72 * DP);
    stageA16x128(at, t + ((size_t)(b * DP + k) * NL + lt * 16) * DP, DP);
    __syncthreads();

    int lane = threadIdx.x & 31;
    int mt   = mtg * 8 + (threadIdx.x >> 5);
    const unsigned short* arow = at + (lane & 15) * APAD;
    const unsigned short* brow = vj + ((size_t)(b * NL + mt * 16 + (lane & 15)) * DP);
    v8f c0 = {}, c1 = {};
    gemmK128(arow, brow, lane, c0, c1);
    float* out = feat + ((size_t)(b * DP + k) * NL + lt * 16) * NL + mt * 16 + (lane & 15);
    int mb = (lane >> 4) * 8;
    for (int r = 0; r < 8; ++r) out[(size_t)(mb + r) * NL] = c0[r] + c1[r];
}

// ---------------- stats: deterministic per-(b,c) sum / sumsq ----------------
__global__ __launch_bounds__(256) void stats_kernel(
    const float* __restrict__ feat, float* __restrict__ stats) {
    __shared__ float s1s[256], s2s[256];
    int bk = blockIdx.x;                       // 0..255 = b*128+c
    const float* p = feat + (size_t)bk * HW;
    float s1 = 0.f, s2 = 0.f;
    for (int i = threadIdx.x; i < HW; i += 256) { float v = p[i]; s1 += v; s2 += v * v; }
    s1s[threadIdx.x] = s1; s2s[threadIdx.x] = s2;
    __syncthreads();
    for (int off = 128; off > 0; off >>= 1) {
        if ((int)threadIdx.x < off) {
            s1s[threadIdx.x] += s1s[threadIdx.x + off];
            s2s[threadIdx.x] += s2s[threadIdx.x + off];
        }
        __syncthreads();
    }
    if (threadIdx.x == 0) { stats[bk * 2] = s1s[0]; stats[bk * 2 + 1] = s2s[0]; }
}

// ---------------- norm + relu + NCHW->NHWC bf16 (vectorized 16B stores) ----------------
__global__ __launch_bounds__(256) void norm_relu_kernel(
    const float* __restrict__ feat, const float* __restrict__ stats,
    const float* __restrict__ gamma, const float* __restrict__ beta,
    unsigned short* __restrict__ fn) {
    __shared__ float sc[DP], sh[DP];
    int b = blockIdx.x / 576;                  // 576 blocks per batch (HW/256)
    if (threadIdx.x < DP) {
        int c = threadIdx.x;
        float n = (float)HW;
        float m   = stats[(b * DP + c) * 2] / n;
        float var = stats[(b * DP + c) * 2 + 1] / n - m * m;
        float s = gamma[c] * rsqrtf(var + 1e-5f);
        sc[c] = s;
        sh[c] = beta[c] - m * s;
    }
    __syncthreads();
    int p   = blockIdx.x * 256 + threadIdx.x;  // global pixel index
    int pix = p % HW;
    const float* src = feat + (size_t)b * DP * HW + pix;
    unsigned short* dst = fn + (size_t)p * DP;
    for (int c = 0; c < DP; c += 8) {
        union { unsigned short u[8]; u32x4 q; } pk;
        for (int e = 0; e < 8; ++e) {
            float v = src[(size_t)(c + e) * HW];
            v = v * sc[c + e] + sh[c + e];
            pk.u[e] = f2bf(v > 0.f ? v : 0.f);
        }
        *(u32x4*)(dst + c) = pk.q;
    }
}

// ---------------- 3x3 conv as implicit GEMM (K = 9*128), +ReLU, NHWC bf16 out ----------
// Block: one (b,y,xt) spatial patch; its 3x18x128 zero-padded halo tile is staged once
// into LDS and shared by 8 waves covering ot = 0..7.
__global__ __launch_bounds__(256) void conv3_kernel(
    const unsigned short* __restrict__ fn, const unsigned short* __restrict__ wB,
    unsigned short* __restrict__ fn2) {
    __shared__ __align__(16) unsigned short tile[54 * APAD];   // 3 rows x 18 px, padded
    int bid = blockIdx.x;                  // 0..18431
    int xt = bid % 24;
    int y  = (bid / 24) % NL;
    int b  = bid / (24 * NL);

    // cooperative halo stage: 54 pixels x 16 chunks = 864 16B-moves over 256 threads
    for (int id = threadIdx.x; id < 54 * 16; id += 256) {
        int cc  = (id & 15) * 8;
        int pix = id >> 4;                 // 0..53
        int row = pix / 18, px = pix % 18;
        int yy = y + row - 1;
        int xx = xt * 16 + px - 1;
        u32x4 val = {0u, 0u, 0u, 0u};
        if (yy >= 0 && yy < NL && xx >= 0 && xx < NL)
            val = *(const u32x4*)(fn + ((size_t)(b * NL + yy) * NL + xx) * DP + cc);
        *(u32x4*)(tile + pix * APAD + cc) = val;
    }
    __syncthreads();

    int lane = threadIdx.x & 31;
    int ot   = threadIdx.x >> 5;
    const unsigned short* brow = wB + (ot * 16 + (lane & 15)) * 1152;
    v8f c0 = {}, c1 = {};                  // dual accumulators: WMMA RAW chain 18 deep, not 36
    for (int r = 0; r < 9; ++r) {
        if (r < 8) __builtin_prefetch((const void*)(brow + (r + 1) * DP), 0, 1);
        int pix = (r / 3) * 18 + (lane & 15) + (r % 3);   // halo pixel for this lane's A row
        const unsigned short* arow = tile + pix * APAD;
        for (int ib = 0; ib < DP; ib += 64) {
            BfVec A0, B0, A1, B1;
            loadA(A0, arow + ib, lane);            // ds_load_b128 x2, bank-conflict-free
            loadB(B0, brow + r * DP + ib, lane);
            loadA(A1, arow + ib + 32, lane);
            loadB(B1, brow + r * DP + ib + 32, lane);
            c0 = wmma_bf16(A0, B0, c0);
            c1 = wmma_bf16(A1, B1, c1);
        }
    }
    unsigned short* out = fn2 + ((size_t)(b * NL + y) * NL + xt * 16) * DP + ot * 16 + (lane & 15);
    int mb = (lane >> 4) * 8;
    for (int rr = 0; rr < 8; ++rr) {
        float v = c0[rr] + c1[rr];
        out[(size_t)(mb + rr) * DP] = f2bf(v > 0.f ? v : 0.f);
    }
}

// ---------------- 1x1 conv (K=128) + bias, fp32 NCHW out ----------------
// Block: one (b,y,xt) pixel-row tile staged in LDS; 8 waves cover ot = 0..7.
__global__ __launch_bounds__(256) void conv1_kernel(
    const unsigned short* __restrict__ fn2, const unsigned short* __restrict__ w1,
    const float* __restrict__ outb, float* __restrict__ out) {
    __shared__ __align__(16) unsigned short at[16 * APAD];
    int bid = blockIdx.x;                  // 0..18431
    int xt = bid % 24;
    int y  = (bid / 24) % NL;
    int b  = bid / (24 * NL);
    stageA16x128(at, fn2 + ((size_t)(b * NL + y) * NL + xt * 16) * DP, DP);
    __syncthreads();

    int lane = threadIdx.x & 31;
    int ot   = threadIdx.x >> 5;
    const unsigned short* arow = at + (lane & 15) * APAD;
    const unsigned short* brow = w1 + (ot * 16 + (lane & 15)) * DP;
    v8f c0 = {}, c1 = {};
    gemmK128(arow, brow, lane, c0, c1);
    int o = ot * 16 + (lane & 15);
    float bias = outb[o];
    float* dst = out + ((size_t)(b * DP + o) * NL + y) * NL + xt * 16;
    int mb = (lane >> 4) * 8;
    for (int rr = 0; rr < 8; ++rr) dst[mb + rr] = c0[rr] + c1[rr] + bias;
}

// ---------------- launcher ----------------
extern "C" void kernel_launch(void* const* d_in, const int* in_sizes, int n_in,
                              void* d_out, int out_size, void* d_ws, size_t ws_size,
                              hipStream_t stream) {
    const float* h    = (const float*)d_in[0];
    const float* Wi   = (const float*)d_in[1];
    const float* bi   = (const float*)d_in[2];
    const float* Wj   = (const float*)d_in[3];
    const float* bj   = (const float*)d_in[4];
    const float* Wbl  = (const float*)d_in[5];
    const float* gam  = (const float*)d_in[6];
    const float* bet  = (const float*)d_in[7];
    const float* mixw = (const float*)d_in[8];
    const float* outw = (const float*)d_in[9];
    const float* outb = (const float*)d_in[10];
    float* out = (float*)d_out;

    // workspace layout (bytes)
    char* ws = (char*)d_ws;
    size_t off = 0;
    unsigned short* ui_bf = (unsigned short*)(ws + off); off += (size_t)NB * NL * DP * 2;            // 192 KB
    unsigned short* vj_bf = (unsigned short*)(ws + off); off += (size_t)NB * NL * DP * 2;            // 192 KB
    unsigned short* Wb    = (unsigned short*)(ws + off); off += (size_t)DP * DP * DP * 2;            // 4 MB
    unsigned short* tbuf  = (unsigned short*)(ws + off); off += (size_t)NB * DP * NL * DP * 2;       // 25 MB
    float*          feat  = (float*)(ws + off);          size_t feat_off = off;
                                                         off += (size_t)NB * DP * HW * 4;           // 151 MB
    float*          stats = (float*)(ws + off);          off += (size_t)NB * DP * 2 * 4;            // 2 KB
    unsigned short* fn    = (unsigned short*)(ws + off); off += (size_t)NB * HW * DP * 2;           // 75.5 MB
    unsigned short* wmixB = (unsigned short*)(ws + off); off += (size_t)DP * 1152 * 2;              // 288 KB
    unsigned short* w1B   = (unsigned short*)(ws + off); off += (size_t)DP * DP * 2;                // 32 KB
    // fn2 aliases the feat buffer (feat is dead after norm_relu); 75.5 MB < 151 MB
    unsigned short* fn2   = (unsigned short*)(ws + feat_off);

    // stage 0: projections + bf16 repacks (independent)
    proj_kernel<<<NB * NL, 256, 0, stream>>>(h, Wi, bi, Wj, bj, ui_bf, vj_bf);
    repack_bilinear_kernel<<<(DP * DP * DP) / 256, 256, 0, stream>>>(Wbl, Wb);
    repack_mix_kernel<<<(DP * DP * 9) / 256, 256, 0, stream>>>(mixw, wmixB);
    repack_w1_kernel<<<(DP * DP) / 256, 256, 0, stream>>>(outw, w1B);

    // stage 1: t = ui x W   (block = (b,k,lt); waves = jt)
    bilinear1_kernel<<<NB * DP * 24, 256, 0, stream>>>(ui_bf, Wb, tbuf);

    // stage 2: feat = t x vj^T per (b,k)   (block = (b,k,lt,mt-group); waves = mt)
    pairgemm_kernel<<<NB * DP * 24 * 3, 256, 0, stream>>>(tbuf, vj_bf, feat);

    // instance-norm statistics (deterministic reduction), then norm+relu -> NHWC bf16
    stats_kernel<<<NB * DP, 256, 0, stream>>>(feat, stats);
    norm_relu_kernel<<<(NB * HW) / 256, 256, 0, stream>>>(feat, stats, gam, bet, fn);

    // 3x3 conv (implicit GEMM, LDS-staged halo) + relu -> fn2 ; then 1x1 conv + bias -> out
    conv3_kernel<<<NB * NL * 24, 256, 0, stream>>>(fn, wmixB, fn2);
    conv1_kernel<<<NB * NL * 24, 256, 0, stream>>>(fn2, w1B, outb, out);
}
